// ParallelTacotron2Loss_68736656605321
// MI455X (gfx1250) — compile-verified, matrix operand order checked by reading, not run on hardware
//
#include <hip/hip_runtime.h>
#include <hip/hip_bf16.h>

typedef __attribute__((ext_vector_type(16))) _Float16 v16h;
typedef __attribute__((ext_vector_type(8)))  _Float16 h8;
typedef __attribute__((ext_vector_type(8)))  float    v8f;

#define K_ 4
#define B_ 8
#define T_ 800
#define C_ 80
#define CP 96
#define S_ 128
#define Z_ 32
#define KB 32
#define NT 50            // 800/16 tiles per dim
#define GAMMA_ 0.05f
#define INVG 20.0f
#define WARP_ 256.0f
#define BIG_ 100000000.0f

// ---------------- workspace layout (bytes) ----------------
#define OFF_XH  ((size_t)0)                         // 32*800*96 f16 = 4,915,200
#define OFF_YH  ((size_t)4915200)                   //  8*800*96 f16 = 1,228,800
#define OFF_X2  ((size_t)6144000)                   // 32*800 f32   =   102,400
#define OFF_Y2  ((size_t)6246400)                   //  8*800 f32   =    25,600
#define OFF_D   ((size_t)6272000)                   // 32*800*800 f32 = 81,920,000
#define OFF_DTW ((size_t)88192000)                  // 32 f32
#define WS_NEED ((size_t)88192128)

// ---------------- kernel 1: sigmoid -> f16 pack (pad C to 96) + row sq-norms -
__global__ void pt2_sigmoid_pack(const float* __restrict__ mel_iters,
                                 const float* __restrict__ mel_targets,
                                 _Float16* __restrict__ xh, _Float16* __restrict__ yh,
                                 float* __restrict__ x2, float* __restrict__ y2) {
  const int wid  = blockIdx.x * (blockDim.x >> 5) + (threadIdx.x >> 5);
  const int lane = threadIdx.x & 31;
  const int totalX = KB * T_;
  const int totalRows = totalX + B_ * T_;
  if (wid >= totalRows) return;

  const float* src;
  _Float16* dsth;
  float* dstsq;
  if (wid < totalX) {
    src   = mel_iters + (size_t)wid * C_;
    dsth  = xh + (size_t)wid * CP;
    dstsq = x2 + wid;
  } else {
    const int ry = wid - totalX;
    src   = mel_targets + (size_t)ry * C_;
    dsth  = yh + (size_t)ry * CP;
    dstsq = y2 + ry;
  }

  float sq = 0.0f;
  #pragma unroll
  for (int c = lane; c < CP; c += 32) {
    float v = 0.0f;
    if (c < C_) {
      v = 1.0f / (1.0f + __expf(-src[c]));
      sq += v * v;
    }
    dsth[c] = (_Float16)v;
  }
  #pragma unroll
  for (int m = 16; m > 0; m >>= 1) sq += __shfl_xor(sq, m, 32);
  if (lane == 0) *dstsq = sq;
}

// ---------------- kernel 2: batched D = x2 + y2 - 2*X*Y^T via WMMA ----------
__device__ inline v16h load_a16(const _Float16* base, int halfSel) {
  // lane<16 : K chunks [0,8)+[16,24) ; lane>=16 : [8,16)+[24,32)
  const h8 lo = *(const h8*)(base + 8 * halfSel);
  const h8 hi = *(const h8*)(base + 16 + 8 * halfSel);
  v16h v;
  #pragma unroll
  for (int t = 0; t < 8; ++t) { v[t] = lo[t]; v[t + 8] = hi[t]; }
  return v;
}

__global__ void pt2_dist_wmma(const _Float16* __restrict__ xh,
                              const _Float16* __restrict__ yh,
                              const float* __restrict__ x2,
                              const float* __restrict__ y2,
                              float* __restrict__ D) {
  const int wid = blockIdx.x * (blockDim.x >> 5) + (threadIdx.x >> 5);
  if (wid >= KB * NT * NT) return;
  const int bb   = wid / (NT * NT);
  const int tile = wid % (NT * NT);
  const int tm = tile / NT, tn = tile % NT;
  const int lane = threadIdx.x & 31;
  const int halfSel = lane >> 4;
  const int r = lane & 15;

  const _Float16* Xb = xh + (size_t)bb * T_ * CP;
  const _Float16* Yb = yh + (size_t)(bb % B_) * T_ * CP;

  v8f acc = {};
  #pragma unroll
  for (int kk = 0; kk < 3; ++kk) {
    const int k0 = kk * 32;
    v16h a = load_a16(Xb + (size_t)(tm * 16 + r) * CP + k0, halfSel);
    v16h b = load_a16(Yb + (size_t)(tn * 16 + r) * CP + k0, halfSel);
    acc = __builtin_amdgcn_wmma_f32_16x16x32_f16(false, a, false, b,
                                                 (short)0, acc, false, false);
  }

  const float* x2b = x2 + bb * T_;
  const float  yn  = y2[(bb % B_) * T_ + tn * 16 + r];
  float* Db = D + (size_t)bb * T_ * T_;
  const int n = tn * 16 + r;
  #pragma unroll
  for (int t = 0; t < 8; ++t) {
    const int m = tm * 16 + t + 8 * halfSel;     // C/D layout: VGPR t -> M = t + 8*half
    Db[(size_t)m * T_ + n] = x2b[m] + yn - 2.0f * acc[t];
  }
}

// ---------------- kernel 3: soft-DTW anti-diagonal wavefront ----------------
__global__ __launch_bounds__(800)
void pt2_sdtw(const float* __restrict__ D, float* __restrict__ dtw) {
  __shared__ float buf[3 * (T_ + 1)];
  const int bb  = blockIdx.x;
  const int tid = threadIdx.x;          // 0..799 -> i = tid+1
  const float* Db = D + (size_t)bb * T_ * T_;

  for (int idx = tid; idx <= T_; idx += T_) {
    buf[idx]          = (idx == 0) ? 0.0f : BIG_;   // "two" (d-2) slot 0
    buf[(T_ + 1) + idx] = BIG_;                     // "one" (d-1) slot 1
  }
  __syncthreads();

  const int i = tid + 1;
  // software prefetch of D for d=2
  float dval = 0.0f;
  {
    const int j = 2 - i;
    if (j >= 1 && j <= T_) dval = Db[(size_t)(i - 1) * T_ + (j - 1)];
  }

  for (int d = 2; d <= T_ + T_; ++d) {
    float* cur        = buf + (size_t)(T_ + 1) * (d % 3);
    const float* one  = buf + (size_t)(T_ + 1) * ((d - 1) % 3);
    const float* two  = buf + (size_t)(T_ + 1) * ((d - 2) % 3);
    const int j = d - i;
    float v = BIG_;
    if (j >= 1 && j <= T_) {
      const float diag = two[i - 1];
      const float up   = one[i - 1] + WARP_;
      const float left = one[i] + WARP_;
      const float mn = fminf(diag, fminf(up, left));
      const float s = __expf((mn - diag) * INVG) + __expf((mn - up) * INVG) +
                      __expf((mn - left) * INVG);
      v = dval + (mn - GAMMA_ * __logf(s));
    }
    cur[i] = v;
    if (tid == 0) cur[0] = BIG_;
    // prefetch next diagonal's D element (independent of LDS state)
    const int jn = d + 1 - i;
    if (jn >= 1 && jn <= T_) dval = Db[(size_t)(i - 1) * T_ + (jn - 1)];
    __syncthreads();
  }
  if (tid == 0) dtw[bb] = buf[(size_t)(T_ + 1) * ((2 * T_) % 3) + T_];
}

// ---------------- kernel 4: epilogue (losses, KL, anneal) -------------------
__global__ void pt2_final(const float* __restrict__ dtw,
                          const int* __restrict__ mel_lens,
                          const int* __restrict__ src_lens,
                          const float* __restrict__ durations,
                          const float* __restrict__ mus,
                          const float* __restrict__ log_vars,
                          const int* __restrict__ step,
                          float* __restrict__ out) {
  __shared__ float red[256];
  const int t = threadIdx.x;
  float p = 0.0f;
  for (int idx = t; idx < B_ * Z_; idx += 256) {
    const float mu = mus[idx], lv = log_vars[idx];
    p += 1.0f + lv - mu * mu - __expf(lv);
  }
  red[t] = p;
  __syncthreads();
  for (int s2 = 128; s2 > 0; s2 >>= 1) {
    if (t < s2) red[t] += red[t + s2];
    __syncthreads();
  }
  if (t == 0) {
    const float kl = -0.5f * red[0];
    float mel_iter = 0.0f;
    for (int kb = 0; kb < KB; ++kb) mel_iter += dtw[kb];
    mel_iter *= (1.0f / B_);                       // sum_k mean_b
    float mel = 0.0f, dur = 0.0f;
    for (int b = 0; b < B_; ++b) {
      const float lenf = (float)mel_lens[b];
      mel += mel_iter / ((float)K_ * lenf);
      float sd = 0.0f;
      for (int s = 0; s < S_; ++s) sd += durations[b * S_ + s];
      dur += fabsf(sd - lenf) / (float)src_lens[b];
    }
    mel *= (1.0f / B_);
    dur = 2.0f * dur * (1.0f / B_);                // LAMBDA * mean
    const int st = step[0];
    float beta;
    if (st < 2000)      beta = 0.0f;
    else if (st >= 8000) beta = 1.0f;
    else                 beta = (float)(st - 2000) / 6000.0f;
    out[0] = mel + dur + beta * kl;
    out[1] = mel;
    out[2] = dur;
    out[3] = kl;
    out[4] = beta;
  }
}

extern "C" void kernel_launch(void* const* d_in, const int* in_sizes, int n_in,
                              void* d_out, int out_size, void* d_ws, size_t ws_size,
                              hipStream_t stream) {
  if (ws_size < WS_NEED) return;  // workspace too small; deterministic no-op
  const float* mel_iters   = (const float*)d_in[0];
  const float* mel_targets = (const float*)d_in[1];
  const int*   mel_lens    = (const int*)d_in[2];
  const int*   src_lens    = (const int*)d_in[3];
  const float* durations   = (const float*)d_in[4];
  const float* mus         = (const float*)d_in[5];
  const float* log_vars    = (const float*)d_in[6];
  const int*   step        = (const int*)d_in[7];
  float* out = (float*)d_out;

  char* ws = (char*)d_ws;
  _Float16* xh = (_Float16*)(ws + OFF_XH);
  _Float16* yh = (_Float16*)(ws + OFF_YH);
  float* x2  = (float*)(ws + OFF_X2);
  float* y2  = (float*)(ws + OFF_Y2);
  float* D   = (float*)(ws + OFF_D);
  float* dtw = (float*)(ws + OFF_DTW);

  // 1) sigmoid + f16 pack + norms: (32+8)*800 rows, 1 wave/row, 8 waves/block
  {
    const int rows = (KB + B_) * T_;
    const int blocks = (rows + 7) / 8;
    pt2_sigmoid_pack<<<blocks, 256, 0, stream>>>(mel_iters, mel_targets, xh, yh, x2, y2);
  }
  // 2) batched distance GEMM: 32 * 50 * 50 tiles, 1 wave/tile, 8 waves/block
  {
    const int tiles = KB * NT * NT;
    const int blocks = (tiles + 7) / 8;
    pt2_dist_wmma<<<blocks, 256, 0, stream>>>(xh, yh, x2, y2, D);
  }
  // 3) soft-DTW wavefront: 1 block / batch, 800 threads (25 wave32s)
  pt2_sdtw<<<KB, 800, 0, stream>>>(D, dtw);
  // 4) scalar epilogue
  pt2_final<<<1, 256, 0, stream>>>(dtw, mel_lens, src_lens, durations, mus,
                                   log_vars, step, out);
}
